// SoftAlignGlobalAttention_23373212025421
// MI455X (gfx1250) — compile-verified, matrix operand order checked by reading, not run on hardware
//
#include <hip/hip_runtime.h>

typedef _Float16 v16h __attribute__((ext_vector_type(16)));
typedef _Float16 half8 __attribute__((ext_vector_type(8)));
typedef float    v8f  __attribute__((ext_vector_type(8)));

#define WMMA_F16(a, b, c) \
    __builtin_amdgcn_wmma_f32_16x16x32_f16(false, (a), false, (b), (short)0, (c), false, false)

// Compiler-only memory fence. Cross-lane LDS exchange within one wave is safe
// with just this: the ISA guarantees same-wave LDS ops are processed in order
// (DScnt in-order), and register deps get s_wait_dscnt from the compiler.
#define WAVE_MEM_FENCE() asm volatile("" ::: "memory")

// ---------------------------------------------------------------------------
// Fragment loader: A-matrix 16x32 f16 layout (ISA 7.12.2).
//   lanes 0-15 : row = lane,    halves = K{0..7}  and K{16..23}
//   lanes 16-31: row = lane-16, halves = K{8..15} and K{24..31}
// B fragments (K x 16) mirror this with the lane indexing the column.
// ---------------------------------------------------------------------------
static __device__ inline v16h load_frag16(const _Float16* rowbase, int lane) {
    const int off = (lane & 16) ? 8 : 0;
    union { v16h v; half8 h[2]; } u;
    u.h[0] = *reinterpret_cast<const half8*>(rowbase + off);
    u.h[1] = *reinterpret_cast<const half8*>(rowbase + off + 16);
    return u.v;
}

// ---------------------------------------------------------------------------
// f32 -> f16 conversion, 8 elements per thread (b128 in, b128 out)
// ---------------------------------------------------------------------------
__global__ void cvt_f32_f16_kernel(const float4* __restrict__ in,
                                   half8* __restrict__ out, int n8) {
    int i = blockIdx.x * blockDim.x + threadIdx.x;
    if (i < n8) {
        float4 a = in[i * 2];
        float4 b = in[i * 2 + 1];
        half8 h = {(_Float16)a.x, (_Float16)a.y, (_Float16)a.z, (_Float16)a.w,
                   (_Float16)b.x, (_Float16)b.y, (_Float16)b.z, (_Float16)b.w};
        out[i] = h;
    }
}

// ---------------------------------------------------------------------------
// GEMM: out[m,n] = sum_k A[m,k] * W[n,k] + bias[n]   (x @ W^T + b)
// Per wave: 32x64 output tile (2 M-subtiles x 4 N-subtiles = 8 accumulators).
// Ping-pong pipelined K loop in steps of 64 (two 32-deep fragment sets in
// distinct registers -> no rotation moves, partial loadcnt waits).
// Requires K % 64 == 0, K >= 64.
// mode 0: f32 store row-major           (final output projection)
// mode 1: RoPE epilogue, f16 row-major  (Q and K projections)
// mode 2: f16 store transposed to [b, h, d, T] (V projection)
// ---------------------------------------------------------------------------
__global__ __launch_bounds__(256) void gemm_wmma_kernel(
    const _Float16* __restrict__ A, const _Float16* __restrict__ W,
    const float* __restrict__ bias, void* __restrict__ out,
    int M, int N, int K, int mode, int T, int HD) {

    const int lane = threadIdx.x & 31;
    const int wave = threadIdx.x >> 5;
    const int w    = blockIdx.x * 8 + wave;
    const int ntiles = N >> 6;
    const int m0 = (w / ntiles) << 5;
    const int n0 = (w % ntiles) << 6;
    if (m0 >= M) return;

    v8f acc[2][4] = {};

    const _Float16* aptr0 = A + (size_t)(m0 + (lane & 15)) * K;
    const _Float16* aptr1 = A + (size_t)(m0 + 16 + (lane & 15)) * K;
    const _Float16* bptr[4];
#pragma unroll
    for (int nt = 0; nt < 4; ++nt)
        bptr[nt] = W + (size_t)(n0 + nt * 16 + (lane & 15)) * K;

    auto loadset = [&](int k0, v16h& a0, v16h& a1, v16h* bf) {
        a0 = load_frag16(aptr0 + k0, lane);
        a1 = load_frag16(aptr1 + k0, lane);
#pragma unroll
        for (int nt = 0; nt < 4; ++nt)
            bf[nt] = load_frag16(bptr[nt] + k0, lane);
    };
    auto mma = [&](const v16h& a0, const v16h& a1, const v16h* bf) {
#pragma unroll
        for (int nt = 0; nt < 4; ++nt) {
            acc[0][nt] = WMMA_F16(a0, bf[nt], acc[0][nt]);
            acc[1][nt] = WMMA_F16(a1, bf[nt], acc[1][nt]);
        }
    };

    v16h a0A, a1A, bA[4];     // ping
    v16h a0B, a1B, bB[4];     // pong
    loadset(0, a0A, a1A, bA);
#pragma unroll 1
    for (int k0 = 0; k0 < K - 64; k0 += 64) {
        loadset(k0 + 32, a0B, a1B, bB);
        mma(a0A, a1A, bA);
        loadset(k0 + 64, a0A, a1A, bA);
        mma(a0B, a1B, bB);
    }
    loadset(K - 32, a0B, a1B, bB);
    mma(a0A, a1A, bA);
    mma(a0B, a1B, bB);

    // Epilogue. C/D layout: VGPR j -> row base + j (+8 for lanes>=16), col = lane&15.
#pragma unroll
    for (int mt = 0; mt < 2; ++mt) {
        const int rbase = m0 + mt * 16 + ((lane & 16) >> 1);
#pragma unroll
        for (int nt = 0; nt < 4; ++nt) {
            const int col = n0 + nt * 16 + (lane & 15);
            const float bv = bias[col];
#pragma unroll
            for (int j = 0; j < 8; ++j) {
                float v = acc[mt][nt][j] + bv;
                const int row = rbase + j;
                if (mode == 0) {
                    ((float*)out)[(size_t)row * N + col] = v;
                } else if (mode == 1) {
                    // RoPE: adjacent channels sit in adjacent lanes.
                    float partner = __shfl_xor(v, 1, 32);
                    const int d = col & (HD - 1);
                    const float invf =
                        __powf(10000.0f, -(float)(d & ~1) / (float)HD);
                    const float ang = (float)(row % T) * invf;
                    float s, c;
                    sincosf(ang, &s, &c);
                    const float r = (col & 1) ? (partner * s + v * c)
                                              : (v * c - partner * s);
                    ((_Float16*)out)[(size_t)row * N + col] = (_Float16)r;
                } else {
                    const int bb = row / T, t = row % T;
                    const int hh = col / HD, d = col & (HD - 1);
                    const size_t idx =
                        ((size_t)((bb * (N / HD) + hh) * HD + d)) * T + t;
                    ((_Float16*)out)[idx] = (_Float16)v;
                }
            }
        }
    }
}

// ---------------------------------------------------------------------------
// Flash attention, one wave per 16 query rows of one (b,h).
// q,k: [B,T,C] f16 (post-RoPE), vt: [B,H,HD,T] f16, ao: [B,T,C] f16.
// Each wave uses a private LDS slice to re-layout P from C-layout to A-layout;
// no inter-wave barriers needed (same-wave LDS is in-order).
// V-fragment loads are issued right after the S WMMAs so their latency hides
// behind the online-softmax VALU work.
// ---------------------------------------------------------------------------
__global__ __launch_bounds__(256) void attn_wmma_kernel(
    const _Float16* __restrict__ qb, const _Float16* __restrict__ kb,
    const _Float16* __restrict__ vt, _Float16* __restrict__ ao,
    int H, int T, int HD) {

    __shared__ __align__(16) _Float16 plds[8 * 16 * 32];

    const int lane = threadIdx.x & 31;
    const int l15  = lane & 15;
    const int wave = threadIdx.x >> 5;
    _Float16* myP = plds + wave * 16 * 32;

    const int w = blockIdx.x * 8 + wave;
    const int qtiles = T >> 4;
    const int qt = w % qtiles;
    const int h  = (w / qtiles) % H;
    const int b  = w / (qtiles * H);
    const int C  = H * HD;
    const int q0 = qt << 4;

    // Q A-fragments (16 x 64 split into two 16x32)
    const _Float16* qrow = qb + (size_t)(b * T + q0 + l15) * C + h * HD;
    const v16h qa0 = load_frag16(qrow, lane);
    const v16h qa1 = load_frag16(qrow + 32, lane);

    v8f acc[4] = {};
    float mstat[8], lstat[8];
#pragma unroll
    for (int j = 0; j < 8; ++j) { mstat[j] = -3.0e38f; lstat[j] = 0.0f; }

    const float scale = rsqrtf((float)HD);
    const _Float16* kbase = kb + (size_t)(b * T) * C + h * HD;
    const _Float16* vbase = vt + (size_t)(b * H + h) * HD * T;

#pragma unroll 1
    for (int kc = 0; kc < T; kc += 32) {
        // Prefetch next key/value tiles into cache while we do softmax VALU work.
        if (kc + 32 < T) {
            __builtin_prefetch(kbase + (size_t)(kc + 32 + l15) * C, 0, 0);
            __builtin_prefetch(kbase + (size_t)(kc + 48 + l15) * C, 0, 0);
            __builtin_prefetch(vbase + (size_t)lane * T + kc + 32, 0, 0);
            __builtin_prefetch(vbase + (size_t)(32 + lane) * T + kc + 32, 0, 0);
        }

        // --- all four K fragments first, then both WMMA pairs ---
        const _Float16* kr0 = kbase + (size_t)(kc + l15) * C;
        const _Float16* kr1 = kbase + (size_t)(kc + 16 + l15) * C;
        v16h kf0 = load_frag16(kr0, lane);
        v16h kf1 = load_frag16(kr0 + 32, lane);
        v16h kf2 = load_frag16(kr1, lane);
        v16h kf3 = load_frag16(kr1 + 32, lane);

        v8f s0 = {}, s1 = {};
        s0 = WMMA_F16(qa0, kf0, s0);
        s0 = WMMA_F16(qa1, kf1, s0);
        s1 = WMMA_F16(qa0, kf2, s1);
        s1 = WMMA_F16(qa1, kf3, s1);

        // Issue V-fragment loads early; consumed only after the softmax below.
        v16h vb[4];
#pragma unroll
        for (int t2 = 0; t2 < 4; ++t2) {
            const _Float16* vr = vbase + (size_t)(t2 * 16 + l15) * T + kc;
            vb[t2] = load_frag16(vr, lane);
        }

        // --- online softmax per row (row lives in one 16-lane half) ---
        WAVE_MEM_FENCE();
#pragma unroll
        for (int j = 0; j < 8; ++j) {
            float a = s0[j] * scale;
            float c = s1[j] * scale;
            float mx = fmaxf(a, c);
            mx = fmaxf(mx, __shfl_xor(mx, 1, 32));
            mx = fmaxf(mx, __shfl_xor(mx, 2, 32));
            mx = fmaxf(mx, __shfl_xor(mx, 4, 32));
            mx = fmaxf(mx, __shfl_xor(mx, 8, 32));
            const float nm = fmaxf(mstat[j], mx);
            const float al = __expf(mstat[j] - nm);
            const float p0 = __expf(a - nm);
            const float p1 = __expf(c - nm);
            float rs = p0 + p1;
            rs += __shfl_xor(rs, 1, 32);
            rs += __shfl_xor(rs, 2, 32);
            rs += __shfl_xor(rs, 4, 32);
            rs += __shfl_xor(rs, 8, 32);
            lstat[j] = lstat[j] * al + rs;
            mstat[j] = nm;
#pragma unroll
            for (int t2 = 0; t2 < 4; ++t2) acc[t2][j] *= al;

            // stage P (C-layout) into LDS row-major [16 q x 32 keys]
            const int row = j + ((lane & 16) >> 1);
            myP[row * 32 + l15]      = (_Float16)p0;
            myP[row * 32 + 16 + l15] = (_Float16)p1;
        }
        WAVE_MEM_FENCE();

        // --- reload P as A-fragment, multiply with V ---
        v16h pa = load_frag16(myP + l15 * 32, lane);
#pragma unroll
        for (int t2 = 0; t2 < 4; ++t2)
            acc[t2] = WMMA_F16(pa, vb[t2], acc[t2]);
        WAVE_MEM_FENCE();
    }

    // --- normalize and store [B,T,C] f16 ---
#pragma unroll
    for (int t2 = 0; t2 < 4; ++t2) {
#pragma unroll
        for (int j = 0; j < 8; ++j) {
            const int row = q0 + j + ((lane & 16) >> 1);
            const float v = acc[t2][j] / lstat[j];
            ao[(size_t)(b * T + row) * C + h * HD + t2 * 16 + l15] =
                (_Float16)v;
        }
    }
}

// ---------------------------------------------------------------------------
// Host launcher
// ---------------------------------------------------------------------------
extern "C" void kernel_launch(void* const* d_in, const int* in_sizes, int n_in,
                              void* d_out, int out_size, void* d_ws, size_t ws_size,
                              hipStream_t stream) {
    const float* x    = (const float*)d_in[0];
    const float* cond = (const float*)d_in[1];
    const float* Wq   = (const float*)d_in[2];
    const float* bq   = (const float*)d_in[3];
    const float* Wk   = (const float*)d_in[4];
    const float* bk   = (const float*)d_in[5];
    const float* Wv   = (const float*)d_in[6];
    const float* bv   = (const float*)d_in[7];
    const float* Wo   = (const float*)d_in[8];
    const float* bo   = (const float*)d_in[9];

    const int B = 2, T = 2048, C = 1024, H = 16, HD = 64;
    const int M = B * T;                 // 4096
    const size_t actN = (size_t)M * C;   // 4 Mi elements
    const size_t wN   = (size_t)C * C;   // 1 Mi elements

    _Float16* ws = (_Float16*)d_ws;
    _Float16* xh  = ws;               // [M,C]
    _Float16* ch  = xh  + actN;       // [M,C]
    _Float16* qbf = ch  + actN;       // [M,C]  q post-rope
    _Float16* kbf = qbf + actN;       // [M,C]  k post-rope
    _Float16* vtb = kbf + actN;       // [B,H,HD,T]
    _Float16* aob = vtb + actN;       // [M,C]  attention output
    _Float16* wqh = aob + actN;
    _Float16* wkh = wqh + wN;
    _Float16* wvh = wkh + wN;
    _Float16* woh = wvh + wN;

    const int cvtBlk = 256;
    const int actB = (int)(actN / 8 + cvtBlk - 1) / cvtBlk;
    const int wB   = (int)(wN / 8 + cvtBlk - 1) / cvtBlk;
    cvt_f32_f16_kernel<<<actB, cvtBlk, 0, stream>>>((const float4*)x,    (half8*)xh,  (int)(actN / 8));
    cvt_f32_f16_kernel<<<actB, cvtBlk, 0, stream>>>((const float4*)cond, (half8*)ch,  (int)(actN / 8));
    cvt_f32_f16_kernel<<<wB,   cvtBlk, 0, stream>>>((const float4*)Wq,   (half8*)wqh, (int)(wN / 8));
    cvt_f32_f16_kernel<<<wB,   cvtBlk, 0, stream>>>((const float4*)Wk,   (half8*)wkh, (int)(wN / 8));
    cvt_f32_f16_kernel<<<wB,   cvtBlk, 0, stream>>>((const float4*)Wv,   (half8*)wvh, (int)(wN / 8));
    cvt_f32_f16_kernel<<<wB,   cvtBlk, 0, stream>>>((const float4*)Wo,   (half8*)woh, (int)(wN / 8));

    // GEMM grid: (M/32)*(C/64) = 2048 waves, 8 waves per 256-thread block.
    const int gemmBlocks = (M / 32) * (C / 64) / 8;   // 256
    gemm_wmma_kernel<<<gemmBlocks, 256, 0, stream>>>(xh, wqh, bq, qbf, M, C, C, 1, T, HD);
    gemm_wmma_kernel<<<gemmBlocks, 256, 0, stream>>>(ch, wkh, bk, kbf, M, C, C, 1, T, HD);
    gemm_wmma_kernel<<<gemmBlocks, 256, 0, stream>>>(ch, wvh, bv, vtb, M, C, C, 2, T, HD);

    // Attention: B*H*(T/16) = 4096 waves, 8 waves per 256-thread block.
    const int attnBlocks = B * H * (T / 16) / 8;      // 512
    attn_wmma_kernel<<<attnBlocks, 256, 0, stream>>>(qbf, kbf, vtb, aob, H, T, HD);

    // Output projection -> f32 d_out
    gemm_wmma_kernel<<<gemmBlocks, 256, 0, stream>>>(aob, woh, bo, d_out, M, C, C, 0, T, HD);
}
